// FCSelfAttention_42537356099842
// MI455X (gfx1250) — matrix operand order repaired and checked by reading, not compiled, not run on hardware
//
#include <hip/hip_runtime.h>
#include <hip/hip_bf16.h>

typedef __bf16 bf16;
typedef __attribute__((ext_vector_type(16))) __bf16 v16bf;
typedef __attribute__((ext_vector_type(8)))  float  v8f;
typedef __attribute__((ext_vector_type(4))) unsigned uint32x4;
typedef __attribute__((ext_vector_type(8))) int      int32x8;
typedef __attribute__((ext_vector_type(4))) int      int32x4;

#define B_      4
#define N_      2048
#define DIM_    512
#define H_      8
#define DH_     64
#define INNER_  512
#define QKVC_   1536
#define SCALE_  0.125f
#define NEG_BIG (-3.0e38f)

#if __has_builtin(__builtin_amdgcn_tensor_load_to_lds) && \
    __has_builtin(__builtin_amdgcn_s_wait_tensorcnt)
#define USE_TDM 1
#else
#define USE_TDM 0
#endif

union FragAB { v16bf v; uint4 q[2]; bf16 e[16]; };
union Pack8  { uint4 q; bf16 e[8]; };

#if USE_TDM
// ---------------------------------------------------------------------------
// Tensor Data Mover: 2D tile (rows x cols bf16, row stride in elements) from
// global memory into LDS at lds_off. D# built per CDNA5 ISA §8:
//   group0: count=1 | lds_addr | global_addr[56:0] | type=2
//   group1: data_size=1(2B) | tensor_dim0/1 | tile_dim0/1 | dim0_stride
// Wave-level op: EXEC ignored; tracked by TENSORcnt.
// 6-arg builtin form (clang-23 / amdgpu-toolchain): (g0, g1, g2, g3, g4, cpol)
// ---------------------------------------------------------------------------
__device__ __forceinline__ void tdm_load_2d(unsigned lds_off, const void* gptr,
                                            unsigned rows, unsigned cols,
                                            unsigned row_stride_elems) {
  unsigned long long ga = (unsigned long long)(uintptr_t)gptr;
  uint32x4 g0;
  g0[0] = 1u;                                     // count=1, user descriptor
  g0[1] = lds_off;                                // lds_addr (bytes)
  g0[2] = (unsigned)(ga & 0xFFFFFFFFu);           // global_addr[31:0]
  g0[3] = (unsigned)((ga >> 32) & 0x01FFFFFFu)    // global_addr[56:32]
          | (2u << 30);                           // type = 2 ("image")
  int32x8 g1;
  g1[0] = (int)(1u << 16);                        // data_size = 1 -> 2 bytes
  g1[1] = (int)((cols & 0xFFFFu) << 16);          // tensor_dim0[15:0]
  g1[2] = (int)(((cols >> 16) & 0xFFFFu)          // tensor_dim0[31:16]
          | ((rows & 0xFFFFu) << 16));            // tensor_dim1[15:0]
  g1[3] = (int)(((rows >> 16) & 0xFFFFu)          // tensor_dim1[31:16]
          | ((cols & 0xFFFFu) << 16));            // tile_dim0
  g1[4] = (int)(rows & 0xFFFFu);                  // tile_dim1 (tile_dim2=0)
  g1[5] = (int)row_stride_elems;                  // tensor_dim0_stride[31:0]
  g1[6] = 0;                                      // stride hi / dim1_stride
  g1[7] = 0;
  int32x4 z4 = {0, 0, 0, 0};                      // groups 2/3 unused (2D)
  int32x8 z8 = {0, 0, 0, 0, 0, 0, 0, 0};
  __builtin_amdgcn_tensor_load_to_lds(g0, g1, z4, z4, z8, 0);
}
#endif

// ---------------------------------------------------------------------------
// fp32 -> bf16 conversion (vectorized by 4)
// ---------------------------------------------------------------------------
__global__ __launch_bounds__(256) void f32_to_bf16_kernel(
    const float* __restrict__ src, bf16* __restrict__ dst, int n4) {
  int i = blockIdx.x * blockDim.x + threadIdx.x;
  if (i < n4) {
    float4 f = ((const float4*)src)[i];
    bf16* d = dst + (size_t)i * 4;
    d[0] = (bf16)f.x; d[1] = (bf16)f.y; d[2] = (bf16)f.z; d[3] = (bf16)f.w;
  }
}

// ---------------------------------------------------------------------------
// bf16 GEMM: C[M x Nn] = A[M x K] * Bm[K x Nn], fp32 accumulate via WMMA.
// Block tile 128(M) x 128(N), BK = 32. 8 waves: 2(M) x 4(N); wave tile 64x32
// = 4x2 WMMA tiles -> 8 WMMAs per k-step against 12 ds_load_b128 frag loads.
// A staged row-major; B staged transposed so both fragments are contiguous
// 16B LDS reads matching the ISA 16-bit A/B VGPR layouts.
// ---------------------------------------------------------------------------
template<bool WRITE_F32, bool BIAS>
__global__ __launch_bounds__(256) void gemm_bf16_kernel(
    const bf16* __restrict__ A, const bf16* __restrict__ Bm,
    const float* __restrict__ bias, void* __restrict__ C,
    int M, int Nn, int K)
{
  constexpr int BM = 128, BN = 128, BK = 32;
  __shared__ __align__(16) bf16 As[BM][48];    // 48-elem rows keep 16B align
  __shared__ __align__(16) bf16 Bts[BN][48];

  const int tid  = threadIdx.x;
  const int wave = tid >> 5;
  const int lane = tid & 31;
  const int lan  = lane & 15;
  const int hg   = lane >> 4;          // half-wave group
  const int wm   = wave >> 2;          // 0..1 -> 64 rows
  const int wn   = wave & 3;           // 0..3 -> 32 cols
  const int m0   = blockIdx.y * BM;
  const int n0   = blockIdx.x * BN;

  v8f acc[4][2];
#pragma unroll
  for (int i = 0; i < 4; ++i)
#pragma unroll
    for (int j = 0; j < 2; ++j) acc[i][j] = (v8f)0.0f;

  for (int k0 = 0; k0 < K; k0 += BK) {
    // stage A tile: 128x32, two 16B chunks per thread
#pragma unroll
    for (int s = 0; s < 2; ++s) {
      int c = tid + s * 256;
      int r = c >> 2, cc = (c & 3) * 8;
      *(uint4*)&As[r][cc] =
          *(const uint4*)(A + (size_t)(m0 + r) * K + k0 + cc);
    }
    // stage B tile transposed: 32x128, two 16B chunks per thread
#pragma unroll
    for (int s = 0; s < 2; ++s) {
      int c  = tid + s * 256;
      int kk = c >> 4, cc = (c & 15) * 8;
      Pack8 p;
      p.q = *(const uint4*)(Bm + (size_t)(k0 + kk) * Nn + n0 + cc);
#pragma unroll
      for (int i = 0; i < 8; ++i) Bts[cc + i][kk] = p.e[i];
    }
    if (k0 + BK < K)
      __builtin_prefetch(A + (size_t)(m0 + (tid >> 2)) * K + k0 + BK, 0, 0);
    __syncthreads();

    FragAB aF[4], bF[2];
#pragma unroll
    for (int i = 0; i < 4; ++i) {
      int r = wm * 64 + i * 16 + lan;
      aF[i].q[0] = *(const uint4*)&As[r][hg * 8];
      aF[i].q[1] = *(const uint4*)&As[r][16 + hg * 8];
    }
#pragma unroll
    for (int j = 0; j < 2; ++j) {
      int c = wn * 32 + j * 16 + lan;
      bF[j].q[0] = *(const uint4*)&Bts[c][hg * 8];
      bF[j].q[1] = *(const uint4*)&Bts[c][16 + hg * 8];
    }
#pragma unroll
    for (int i = 0; i < 4; ++i)
#pragma unroll
      for (int j = 0; j < 2; ++j)
        acc[i][j] = __builtin_amdgcn_wmma_f32_16x16x32_bf16(
            false, aF[i].v, false, bF[j].v, (short)0, acc[i][j], false, false);
    __syncthreads();
  }

  // C/D layout: element r of VGPR group -> row = r + hg*8, col = lane&15
#pragma unroll
  for (int i = 0; i < 4; ++i)
#pragma unroll
    for (int j = 0; j < 2; ++j)
#pragma unroll
      for (int r = 0; r < 8; ++r) {
        int row = m0 + wm * 64 + i * 16 + r + hg * 8;
        int col = n0 + wn * 32 + j * 16 + lan;
        float v = acc[i][j][r];
        if constexpr (BIAS) v += bias[col];
        if constexpr (WRITE_F32)
          ((float*)C)[(size_t)row * Nn + col] = v;
        else
          ((bf16*)C)[(size_t)row * Nn + col] = (bf16)v;
      }
}

// ---------------------------------------------------------------------------
// Flash-style attention. Reference semantics: S[n,m] = k_n . q_m * SCALE,
// softmax over m, out[n] = sum_m P[n,m] * v_m.  => K rows are the "queries".
// Block: 128 threads (4 waves); each block owns 64 consecutive n-rows of one
// (b, h). Loop over m in chunks of 64 with online softmax; DH = 64 => two
// 16x16x32 WMMA k-steps per product. K and Q tiles arrive via the Tensor
// Data Mover when available (one DMA per 64x64 tile, TENSORcnt-tracked).
// ---------------------------------------------------------------------------
__global__ __launch_bounds__(128) void attn_kernel(
    const bf16* __restrict__ qkv, bf16* __restrict__ attout)
{
  __shared__ __align__(16) bf16 Ks[64][64];   // K chunk  (A operand of S)
  __shared__ __align__(16) bf16 Qs[64][64];   // Q chunk row-major == Bt for S
  __shared__ __align__(16) bf16 Vts[64][64];  // V chunk transposed == Bt for O
  __shared__ __align__(16) bf16 Ps[64][64];   // P staging (D-layout -> A-layout)

  const int tid  = threadIdx.x;
  const int wave = tid >> 5;          // 0..3 -> 16 n-rows each
  const int lane = tid & 31;
  const int lan  = lane & 15;
  const int hg   = lane >> 4;
  const int bh   = blockIdx.x;
  const int b    = bh >> 3;
  const int h    = bh & 7;
  const int n0   = blockIdx.y * 64;

  const bf16* qbase = qkv + (size_t)(b * N_) * QKVC_ + h * DH_;
  const bf16* kbase = qbase + INNER_;
  const bf16* vbase = qbase + 2 * INNER_;

  // load this block's K rows once (64x64 bf16)
#if USE_TDM
  if (tid < 32)
    tdm_load_2d((unsigned)(uintptr_t)&Ks[0][0],
                kbase + (size_t)n0 * QKVC_, 64, 64, QKVC_);
#else
#pragma unroll
  for (int s = 0; s < 4; ++s) {
    int c = tid + s * 128;
    int r = c >> 3, cc = (c & 7) * 8;
    *(uint4*)&Ks[r][cc] =
        *(const uint4*)(kbase + (size_t)(n0 + r) * QKVC_ + cc);
  }
#endif

  v8f acc_o[4];
#pragma unroll
  for (int j = 0; j < 4; ++j) acc_o[j] = (v8f)0.0f;
  float mrun[8], lrun[8];
#pragma unroll
  for (int r = 0; r < 8; ++r) { mrun[r] = NEG_BIG; lrun[r] = 0.0f; }

  for (int m0 = 0; m0 < N_; m0 += 64) {
    // stage Q (row-major, TDM) and V (transposed, per-lane) chunks
#if USE_TDM
    if (tid < 32)
      tdm_load_2d((unsigned)(uintptr_t)&Qs[0][0],
                  qbase + (size_t)m0 * QKVC_, 64, 64, QKVC_);
#endif
#pragma unroll
    for (int s = 0; s < 4; ++s) {
      int c = tid + s * 128;
      int r = c >> 3, cc = (c & 7) * 8;
#if !USE_TDM
      *(uint4*)&Qs[r][cc] =
          *(const uint4*)(qbase + (size_t)(m0 + r) * QKVC_ + cc);
#endif
      Pack8 p;
      p.q = *(const uint4*)(vbase + (size_t)(m0 + r) * QKVC_ + cc);
#pragma unroll
      for (int i = 0; i < 8; ++i) Vts[cc + i][r] = p.e[i];
    }
#if USE_TDM
    if (tid < 32) __builtin_amdgcn_s_wait_tensorcnt(0);
#endif
    __syncthreads();

    // S(16x64 per wave) = Ks_wave @ Qs^T
    v8f s_acc[4];
#pragma unroll
    for (int j = 0; j < 4; ++j) s_acc[j] = (v8f)0.0f;
#pragma unroll
    for (int kd = 0; kd < 64; kd += 32) {
      FragAB aF;
      int r = wave * 16 + lan;
      aF.q[0] = *(const uint4*)&Ks[r][kd + hg * 8];
      aF.q[1] = *(const uint4*)&Ks[r][kd + 16 + hg * 8];
#pragma unroll
      for (int j = 0; j < 4; ++j) {
        FragAB bF;
        int c = j * 16 + lan;
        bF.q[0] = *(const uint4*)&Qs[c][kd + hg * 8];
        bF.q[1] = *(const uint4*)&Qs[c][kd + 16 + hg * 8];
        s_acc[j] = __builtin_amdgcn_wmma_f32_16x16x32_bf16(
            false, aF.v, false, bF.v, (short)0, s_acc[j], false, false);
      }
    }

    // online softmax over m: row (r,hg) spans 16 lanes of one half-wave
#pragma unroll
    for (int r = 0; r < 8; ++r) {
      float mx = NEG_BIG;
#pragma unroll
      for (int j = 0; j < 4; ++j) {
        s_acc[j][r] *= SCALE_;
        mx = fmaxf(mx, s_acc[j][r]);
      }
#pragma unroll
      for (int off = 1; off < 16; off <<= 1)
        mx = fmaxf(mx, __shfl_xor(mx, off, 32));
      float mnew  = fmaxf(mrun[r], mx);
      float alpha = __expf(mrun[r] - mnew);
      float rsum  = 0.0f;
#pragma unroll
      for (int j = 0; j < 4; ++j) {
        float p = __expf(s_acc[j][r] - mnew);
        s_acc[j][r] = p;
        rsum += p;
      }
#pragma unroll
      for (int off = 1; off < 16; off <<= 1)
        rsum += __shfl_xor(rsum, off, 32);
      lrun[r] = lrun[r] * alpha + rsum;
      mrun[r] = mnew;
#pragma unroll
      for (int j = 0; j < 4; ++j) acc_o[j][r] *= alpha;
    }

    // D-layout -> A-layout for P via LDS
#pragma unroll
    for (int j = 0; j < 4; ++j)
#pragma unroll
      for (int r = 0; r < 8; ++r)
        Ps[wave * 16 + r + hg * 8][j * 16 + lan] = (bf16)s_acc[j][r];
    __syncthreads();

    // O(16x64 per wave) += P_wave @ V  (Bt = Vts)
#pragma unroll
    for (int kd = 0; kd < 64; kd += 32) {
      FragAB aF;
      int r = wave * 16 + lan;
      aF.q[0] = *(const uint4*)&Ps[r][kd + hg * 8];
      aF.q[1] = *(const uint4*)&Ps[r][kd + 16 + hg * 8];
#pragma unroll
      for (int j = 0; j < 4; ++j) {
        FragAB bF;
        int c = j * 16 + lan;
        bF.q[0] = *(const uint4*)&Vts[c][kd + hg * 8];
        bF.q[1] = *(const uint4*)&Vts[c][kd + 16 + hg * 8];
        acc_o[j] = __builtin_amdgcn_wmma_f32_16x16x32_bf16(
            false, aF.v, false, bF.v, (short)0, acc_o[j], false, false);
      }
    }
    __syncthreads();
  }

  // normalize + store in (b, n, h*DH + d) layout for the output projection
  bf16* outbase = attout + (size_t)(b * N_) * INNER_ + h * DH_;
#pragma unroll
  for (int j = 0; j < 4; ++j)
#pragma unroll
    for (int r = 0; r < 8; ++r) {
      int n = n0 + wave * 16 + r + hg * 8;
      int d = j * 16 + lan;
      outbase[(size_t)n * INNER_ + d] = (bf16)(acc_o[j][r] / lrun[r]);
    }
}

// ---------------------------------------------------------------------------
extern "C" void kernel_launch(void* const* d_in, const int* in_sizes, int n_in,
                              void* d_out, int out_size, void* d_ws, size_t ws_size,
                              hipStream_t stream) {
  (void)in_sizes; (void)n_in; (void)out_size; (void)ws_size;
  const float* X    = (const float*)d_in[0];   // (4, 2048, 512)
  const float* Wqkv = (const float*)d_in[1];   // (512, 1536)
  const float* Wout = (const float*)d_in[2];   // (512, 512)
  const float* bout = (const float*)d_in[3];   // (512,)
  float* out = (float*)d_out;                  // (4, 2048, 512)

  const size_t BN_TOK = (size_t)B_ * N_;       // 8192 tokens
  bf16* Xb    = (bf16*)d_ws;
  bf16* Wqkvb = Xb    + BN_TOK * DIM_;
  bf16* Woutb = Wqkvb + (size_t)DIM_ * QKVC_;
  bf16* QKVb  = Woutb + (size_t)INNER_ * DIM_;
  bf16* AttB  = QKVb  + BN_TOK * QKVC_;

  // fp32 -> bf16 conversions
  {
    int n4 = (int)(BN_TOK * DIM_ / 4);
    f32_to_bf16_kernel<<<n4 / 256, 256, 0, stream>>>(X, Xb, n4);
    n4 = DIM_ * QKVC_ / 4;
    f32_to_bf16_kernel<<<n4 / 256, 256, 0, stream>>>(Wqkv, Wqkvb, n4);
    n4 = INNER_ * DIM_ / 4;
    f32_to_bf16_kernel<<<n4 / 256, 256, 0, stream>>>(Wout, Woutb, n4);
  }

  // QKV projection: (8192 x 512) * (512 x 1536) -> bf16
  gemm_bf16_kernel<false, false>
      <<<dim3(QKVC_ / 128, (int)(BN_TOK / 128)), 256, 0, stream>>>(
          Xb, Wqkvb, nullptr, QKVb, (int)BN_TOK, QKVC_, DIM_);

  // attention: grid (B*H, N/64)
  attn_kernel<<<dim3(B_ * H_, N_ / 64), 128, 0, stream>>>(QKVb, AttB);

  // output projection + bias: (8192 x 512) * (512 x 512) -> fp32
  gemm_bf16_kernel<true, true>
      <<<dim3(DIM_ / 128, (int)(BN_TOK / 128)), 256, 0, stream>>>(
          AttB, Woutb, bout, out, (int)BN_TOK, DIM_, INNER_);
}
